// att0_26_2layer_12910671692342
// MI455X (gfx1250) — matrix operand order repaired and checked by reading, not compiled, not run on hardware
//
#include <hip/hip_runtime.h>
#include <hip/hip_bf16.h>

typedef __attribute__((ext_vector_type(16))) _Float16 v16h;
typedef __attribute__((ext_vector_type(8)))  float    v8f;

#define NEG_SLOPE 0.01f
#define SEQ   50      // tokens per batch (rows of x)
#define DIM   64      // feature dim
#define MPAD  64      // padded M for 16x16 tiling
#define SH    72      // f16 LDS row stride (bank-conflict break)
#define SF    66      // f32 LDS row stride (bank-conflict break)

__device__ __forceinline__ float lrelu(float v) { return fmaxf(v, NEG_SLOPE * v); }

union AFrag { v16h v; unsigned int u[8]; };
union CFrag { v8f  v; float        f[8]; };

// A-fragment: 16x32 f16 tile (rows mrow.., K cols kc..kc+31) from row-major LDS, stride SH.
// ISA 7.12.2: lanes 0-15 = M rows, VGPR0-3 hold K=0..7 pairs, VGPR4-7 K=16..23; lanes 16-31: +8.
__device__ __forceinline__ v16h load_a(const _Float16* lds, int mrow, int kc, int lane) {
  AFrag a;
  const int sel = lane >> 4;
  const _Float16* src = lds + (mrow + (lane & 15)) * SH + kc + sel * 8;
#pragma unroll
  for (int v = 0; v < 8; ++v) {
    const int kb = (v < 4) ? (2 * v) : (16 + 2 * (v - 4));
    a.u[v] = *reinterpret_cast<const unsigned int*>(src + kb);
  }
  return a.v;
}

// B-fragment: 32x16 f16 tile (K rows kr.., N cols ncol..) read from TRANSPOSED weight
// (Wt[n][k] = W[k][n], row-major NxK, stride SH) so each packed K-pair is contiguous.
// lanes 0-15: VGPR0-3 hold K=0..7 pairs, VGPR4-7 K=16..23; lanes 16-31: +8.
__device__ __forceinline__ v16h load_b(const _Float16* wt, int kr, int ncol, int lane) {
  AFrag b;
  const int sel = lane >> 4;
  const _Float16* src = wt + (ncol + (lane & 15)) * SH + kr + sel * 8;
#pragma unroll
  for (int v = 0; v < 8; ++v)
    b.u[v] = *reinterpret_cast<const unsigned int*>(src + ((v < 4) ? (2 * v) : (16 + 2 * (v - 4))));
  return b.v;
}

// One 16x16 f32 output tile of A(64xK=64, f16 LDS) x Wt(64x64 transposed f16 LDS).
__device__ __forceinline__ v8f gemm_tile(const _Float16* aLds, const _Float16* wt,
                                         int mrow, int ncol, int lane) {
  v8f c = {};
#pragma unroll
  for (int kc = 0; kc < DIM; kc += 32) {
    v16h a = load_a(aLds, mrow, kc, lane);
    v16h b = load_b(wt, kc, ncol, lane);
    c = __builtin_amdgcn_wmma_f32_16x16x32_f16(false, a, false, b, (short)0, c, false, false);
  }
  return c;
}

__global__ __launch_bounds__(256) void att2layer_fused(
    const float* __restrict__ emb,
    const float* __restrict__ Wa1, const float* __restrict__ ba1,
    const float* __restrict__ Wa2, const float* __restrict__ ba2,
    const float* __restrict__ W1,  const float* __restrict__ b1,
    const float* __restrict__ W2,  const float* __restrict__ b2,
    float* __restrict__ out)
{
  __shared__ _Float16 s_x  [MPAD * SH];   // x (f16, zero-padded rows 50..63) : GEMM A
  __shared__ _Float16 s_wa1[DIM * SH];    // Wa1^T f16
  __shared__ _Float16 s_wa2[DIM * SH];    // Wa2^T f16
  __shared__ _Float16 s_w1 [DIM * SH];    // W1^T  f16
  __shared__ _Float16 s_w2 [DIM * SH];    // W2^T  f16
  __shared__ _Float16 s_th [MPAD * SH];   // attention raw output (f16, padded) : GEMM A
  __shared__ _Float16 s_a1h[MPAD * SH];   // att_value1 (f16, padded)           : GEMM A
  __shared__ float s_xf [SEQ  * SF];      // x  f32 (attention value + epilogue)
  __shared__ float s_q  [MPAD * SF];      // q f32 (padded; final phase: att2@W2+b2)
  __shared__ float s_k  [MPAD * SF];      // k f32 (padded)
  __shared__ float s_a1f[MPAD * SF];      // att_value1 f32 (padded)
  __shared__ float s_ba1[DIM], s_ba2[DIM], s_b1[DIM], s_b2[DIM], s_e0[DIM];

  const int t    = threadIdx.x;
  const int b    = blockIdx.x;
  const int lane = t & 31;
  const int wave = t >> 5;

  // ---------- Phase 1: stage inputs into LDS ----------
  // zero ONLY the pad rows (50..63) of f16 A-operand arrays (no cross-thread races).
  // Pad rows of later tiles land in padded f32 buffers and are never read (i,j < 50).
  for (int i = t; i < (MPAD - SEQ) * SH; i += 256) {
    s_x  [SEQ * SH + i] = (_Float16)0.f;
    s_th [SEQ * SH + i] = (_Float16)0.f;
    s_a1h[SEQ * SH + i] = (_Float16)0.f;
  }
  for (int i = t; i < DIM * DIM; i += 256) {     // transpose weights -> f16
    const int r = i >> 6, c = i & 63;
    s_wa1[c * SH + r] = (_Float16)Wa1[i];
    s_wa2[c * SH + r] = (_Float16)Wa2[i];
    s_w1 [c * SH + r] = (_Float16)W1[i];
    s_w2 [c * SH + r] = (_Float16)W2[i];
  }
  if (t < DIM) {
    s_ba1[t] = ba1[t]; s_ba2[t] = ba2[t]; s_b1[t] = b1[t]; s_b2[t] = b2[t];
    s_e0[t]  = emb[(size_t)b * 51 * DIM + t];          // embeddings[b, 0, :]
  }
  for (int i = t; i < SEQ * DIM; i += 256) {           // x = embeddings[b, 1:, :]
    const int r = i >> 6, d = i & 63;
    const float v = emb[((size_t)b * 51 + 1 + r) * DIM + d];
    s_xf[r * SF + d] = v;
    s_x [r * SH + d] = (_Float16)v;
  }
  __syncthreads();

  // ---------- Phase 2: q = x@Wa1+ba1, k = x@Wa2+ba2  (32 tiles over 8 waves) ----------
#pragma unroll
  for (int u = 0; u < 4; ++u) {
    const int tt   = wave * 4 + u;
    const bool isK = tt >= 16;
    const int tile = tt & 15;
    const int mrow = (tile >> 2) * 16, ncol = (tile & 3) * 16;
    CFrag c; c.v = gemm_tile(s_x, isK ? s_wa2 : s_wa1, mrow, ncol, lane);
    const int col   = ncol + (lane & 15);
    const int rbase = mrow + ((lane >> 4) << 3);
    const float bb  = (isK ? s_ba2 : s_ba1)[col];
    float* dst      = isK ? s_k : s_q;
#pragma unroll
    for (int r = 0; r < 8; ++r)                  // unconditional: dst is padded
      dst[(rbase + r) * SF + col] = c.f[r] + bb;
  }
  __syncthreads();

  // ---------- Phase 3: attention layer 1 (value = x) ----------
  for (int p = t; p < SEQ * DIM; p += 256) {
    const int i = p >> 6, d = p & 63;
    const float qv = s_q[i * SF + d];
    float m = -1e30f;
    for (int j = 0; j < SEQ; ++j) m = fmaxf(m, lrelu(qv + s_k[j * SF + d]));
    float num = 0.f, den = 0.f;
    for (int j = 0; j < SEQ; ++j) {
      const float e = __expf(lrelu(qv + s_k[j * SF + d]) - m);
      num += e * s_xf[j * SF + d];
      den += e;
    }
    s_th[i * SH + d] = (_Float16)(num / den);
  }
  __syncthreads();

  // ---------- Phase 4: att_value1 = att1 @ W1 + b1 ----------
#pragma unroll
  for (int u = 0; u < 2; ++u) {
    const int tt = wave * 2 + u;
    const int mrow = (tt >> 2) * 16, ncol = (tt & 3) * 16;
    CFrag c; c.v = gemm_tile(s_th, s_w1, mrow, ncol, lane);
    const int col   = ncol + (lane & 15);
    const int rbase = mrow + ((lane >> 4) << 3);
    const float bb  = s_b1[col];
#pragma unroll
    for (int r = 0; r < 8; ++r) {                // unconditional: padded dests
      const float v = c.f[r] + bb;
      s_a1f[(rbase + r) * SF + col] = v;
      s_a1h[(rbase + r) * SH + col] = (_Float16)v;
    }
  }
  __syncthreads();

  // ---------- Phase 5: q2 = a1@Wa1+ba1, k2 = a1@Wa2+ba2 ----------
#pragma unroll
  for (int u = 0; u < 4; ++u) {
    const int tt   = wave * 4 + u;
    const bool isK = tt >= 16;
    const int tile = tt & 15;
    const int mrow = (tile >> 2) * 16, ncol = (tile & 3) * 16;
    CFrag c; c.v = gemm_tile(s_a1h, isK ? s_wa2 : s_wa1, mrow, ncol, lane);
    const int col   = ncol + (lane & 15);
    const int rbase = mrow + ((lane >> 4) << 3);
    const float bb  = (isK ? s_ba2 : s_ba1)[col];
    float* dst      = isK ? s_k : s_q;
#pragma unroll
    for (int r = 0; r < 8; ++r)
      dst[(rbase + r) * SF + col] = c.f[r] + bb;
  }
  __syncthreads();

  // ---------- Phase 6: attention layer 2 (value = att_value1) ----------
  for (int p = t; p < SEQ * DIM; p += 256) {
    const int i = p >> 6, d = p & 63;
    const float qv = s_q[i * SF + d];
    float m = -1e30f;
    for (int j = 0; j < SEQ; ++j) m = fmaxf(m, lrelu(qv + s_k[j * SF + d]));
    float num = 0.f, den = 0.f;
    for (int j = 0; j < SEQ; ++j) {
      const float e = __expf(lrelu(qv + s_k[j * SF + d]) - m);
      num += e * s_a1f[j * SF + d];
      den += e;
    }
    s_th[i * SH + d] = (_Float16)(num / den);
  }
  __syncthreads();

  // ---------- Phase 7: att = att2@W2 + b2 -> s_q (padded, reuse) ----------
#pragma unroll
  for (int u = 0; u < 2; ++u) {
    const int tt = wave * 2 + u;
    const int mrow = (tt >> 2) * 16, ncol = (tt & 3) * 16;
    CFrag c; c.v = gemm_tile(s_th, s_w2, mrow, ncol, lane);
    const int col   = ncol + (lane & 15);
    const int rbase = mrow + ((lane >> 4) << 3);
    const float bb  = s_b2[col];
#pragma unroll
    for (int r = 0; r < 8; ++r)
      s_q[(rbase + r) * SF + col] = c.f[r] + bb;
  }
  __syncthreads();

  // ---------- Phase 8: fused epilogue, fully coalesced global store ----------
  // out = lrelu(e0*x + att + att_value1)
  for (int p = t; p < SEQ * DIM; p += 256) {
    const int r = p >> 6, d = p & 63;
    const float v = s_e0[d] * s_xf[r * SF + d] + s_q[r * SF + d] + s_a1f[r * SF + d];
    out[(size_t)b * (SEQ * DIM) + p] = lrelu(v);
  }
}

extern "C" void kernel_launch(void* const* d_in, const int* in_sizes, int n_in,
                              void* d_out, int out_size, void* d_ws, size_t ws_size,
                              hipStream_t stream) {
  (void)in_sizes; (void)n_in; (void)out_size; (void)d_ws; (void)ws_size;
  const float* emb = (const float*)d_in[0];
  const float* Wa1 = (const float*)d_in[1];
  const float* ba1 = (const float*)d_in[2];
  const float* Wa2 = (const float*)d_in[3];
  const float* ba2 = (const float*)d_in[4];
  const float* W1  = (const float*)d_in[5];
  const float* b1  = (const float*)d_in[6];
  const float* W2  = (const float*)d_in[7];
  const float* b2  = (const float*)d_in[8];
  att2layer_fused<<<512, 256, 0, stream>>>(emb, Wa1, ba1, Wa2, ba2, W1, b1, W2, b2,
                                           (float*)d_out);
}
